// GCN_LSTM_89421219102802
// MI455X (gfx1250) — compile-verified, matrix operand order checked by reading, not run on hardware
//
#include <hip/hip_runtime.h>
#include <hip/hip_bf16.h>
#include <math.h>

// ---------------- model constants ----------------
static constexpr int Bsz = 64, Nn = 64, Fd = 2, Hd = 1024, Ed = 65536;
static constexpr int NT = Bsz * Nn;           // 4096
static constexpr int T_IN = 10, T_OUT = 6;
static constexpr int FRAME = NT * Fd;         // 8192 floats per output frame

typedef __attribute__((ext_vector_type(8)))  float  v8f;
typedef __attribute__((ext_vector_type(16))) __bf16 v16bf;
typedef __attribute__((ext_vector_type(8)))  __bf16 v8bf;

// ---------------- WMMA GEMM: OUT[64,N] = act(A@W^T + bias) ----------------
// A row-major bf16 [64,K]; W row-major bf16 [N,K].  K % 32 == 0, N % 32 == 0.
// Per CDNA5 ISA 7.12.2 (16-bit A 16x32 layout), lane (r=lane&15, h=lane>>4)
// holds k = [kt + 8h, +8) in VGPRs0-3 and k = [kt+16+8h, +8) in VGPRs4-7:
// two contiguous 16B chunks.  B (32x16) mirrors this with n = lane&15.
// Each wave computes a 64x32 output tile (4 M-tiles x 2 N-tiles): per k-chunk
// 12 b128 loads feed 8 WMMAs.  A-tiles are loaded one at a time and consumed
// by both B-tiles immediately to keep live VGPRs ~90 (no accumulator spills).
__device__ __forceinline__ v16bf ld_tile(const __bf16* p) {
  v8bf lo = *(const v8bf*)(p);
  v8bf hi = *(const v8bf*)(p + 16);
  v16bf r;
#pragma unroll
  for (int i = 0; i < 8; ++i) { r[i] = lo[i]; r[i + 8] = hi[i]; }
  return r;
}

__device__ __forceinline__ void gemm_pass(const __bf16* __restrict__ A,
                                          const __bf16* __restrict__ W,
                                          int K, int col0, int row, int koff,
                                          v8f acc0[4], v8f acc1[4]) {
  const __bf16* pb0 = W + (size_t)(col0 + row) * K + koff;       // N-tile 0
  const __bf16* pb1 = W + (size_t)(col0 + 16 + row) * K + koff;  // N-tile 1
  const __bf16* pa  = A + (size_t)(row) * K + koff;
  for (int kt = 0; kt < K; kt += 32) {
    if (kt + 64 <= K) {
      __builtin_prefetch(pb0 + kt + 48, 0, 1);  // global_prefetch_b8 (weight stream)
      __builtin_prefetch(pb1 + kt + 48, 0, 1);
    }
    v16bf b0 = ld_tile(pb0 + kt);
    v16bf b1 = ld_tile(pb1 + kt);
#pragma unroll
    for (int mt = 0; mt < 4; ++mt) {
      v16bf a = ld_tile(pa + (size_t)(mt * 16) * K + kt);
      acc0[mt] = __builtin_amdgcn_wmma_f32_16x16x32_bf16(
          false, a, false, b0, (short)0, acc0[mt], false, false);
      acc1[mt] = __builtin_amdgcn_wmma_f32_16x16x32_bf16(
          false, a, false, b1, (short)0, acc1[mt], false, false);
    }
  }
}

__global__ __launch_bounds__(256)
void k_gemm64(const __bf16* __restrict__ A1, const __bf16* __restrict__ W1, int K1,
              const __bf16* __restrict__ A2, const __bf16* __restrict__ W2, int K2,
              const float* __restrict__ bias, int N, int relu,
              float* __restrict__ outF, __bf16* __restrict__ outB) {
  const int wave = threadIdx.x >> 5;
  const int lane = threadIdx.x & 31;
  const int col0 = (blockIdx.x * 8 + wave) * 32;  // 32 columns per wave
  if (col0 >= N) return;                          // wave-uniform: EXEC stays all-ones
  const int row  = lane & 15;
  const int half = lane >> 4;
  const int koff = half * 8;

  v8f acc0[4], acc1[4];
#pragma unroll
  for (int mt = 0; mt < 4; ++mt)
#pragma unroll
    for (int j = 0; j < 8; ++j) { acc0[mt][j] = 0.0f; acc1[mt][j] = 0.0f; }

  gemm_pass(A1, W1, K1, col0, row, koff, acc0, acc1);
  if (A2) gemm_pass(A2, W2, K2, col0, row, koff, acc0, acc1);

  // C/D layout: n = lane&15 (col), m = half*8 + vgpr_j within each 16-row tile.
  const int n0 = col0 + row, n1 = col0 + 16 + row;
  const float bv0 = bias ? bias[n0] : 0.0f;
  const float bv1 = bias ? bias[n1] : 0.0f;
#pragma unroll
  for (int mt = 0; mt < 4; ++mt) {
#pragma unroll
    for (int j = 0; j < 8; ++j) {
      const int m = mt * 16 + half * 8 + j;
      float v0 = acc0[mt][j] + bv0;
      float v1 = acc1[mt][j] + bv1;
      if (relu) { v0 = fmaxf(v0, 0.0f); v1 = fmaxf(v1, 0.0f); }
      if (outF) { outF[(size_t)m * N + n0] = v0; outF[(size_t)m * N + n1] = v1; }
      if (outB) { outB[(size_t)m * N + n0] = (__bf16)v0; outB[(size_t)m * N + n1] = (__bf16)v1; }
    }
  }
}

// ---------------- elementwise / small kernels ----------------
__global__ void k_f32_to_bf16(const float* __restrict__ s, __bf16* __restrict__ d, int n) {
  for (int i = blockIdx.x * blockDim.x + threadIdx.x; i < n; i += gridDim.x * blockDim.x)
    d[i] = (__bf16)s[i];
}
__global__ void k_fill_f32(float* p, float v, int n) {
  for (int i = blockIdx.x * blockDim.x + threadIdx.x; i < n; i += gridDim.x * blockDim.x)
    p[i] = v;
}
__global__ void k_fill_bf16(__bf16* p, int n) {
  for (int i = blockIdx.x * blockDim.x + threadIdx.x; i < n; i += gridDim.x * blockDim.x)
    p[i] = (__bf16)0.0f;
}
__global__ void k_copy_f32(const float* __restrict__ s, float* __restrict__ d, int n) {
  for (int i = blockIdx.x * blockDim.x + threadIdx.x; i < n; i += gridDim.x * blockDim.x)
    d[i] = s[i];
}
__global__ void k_addvec(const float* a, const float* b, float* d, int n) {
  for (int i = blockIdx.x * blockDim.x + threadIdx.x; i < n; i += gridDim.x * blockDim.x)
    d[i] = a[i] + b[i];
}

// deg[i] initialized to 1 (self loop); count in-degree via atomics
__global__ void k_deg_count(const int* __restrict__ ei, float* __restrict__ deg, int E) {
  int e = blockIdx.x * blockDim.x + threadIdx.x;
  if (e < E) atomicAdd(&deg[ei[E + e]], 1.0f);
}
__global__ void k_dinv(const float* __restrict__ deg, float* __restrict__ dinv, int n) {
  int i = blockIdx.x * blockDim.x + threadIdx.x;
  if (i < n) dinv[i] = rsqrtf(deg[i]);
}
// xw = x @ W^T for F=2 (W row-major [2,2])
__global__ void k_xw(const float* __restrict__ x, const float* __restrict__ W,
                     float* __restrict__ xw, int n) {
  int i = blockIdx.x * blockDim.x + threadIdx.x;
  if (i < n) {
    float x0 = x[i * 2], x1 = x[i * 2 + 1];
    xw[i * 2 + 0] = x0 * W[0] + x1 * W[1];
    xw[i * 2 + 1] = x0 * W[2] + x1 * W[3];
  }
}
// g = xw * dinv^2 + b  (self-loop term, also serves as scatter accumulator init)
__global__ void k_self_init(const float* __restrict__ xw, const float* __restrict__ dinv,
                            const float* __restrict__ bg, float* __restrict__ g, int n) {
  int i = blockIdx.x * blockDim.x + threadIdx.x;
  if (i < n) {
    float w = dinv[i] * dinv[i];
    g[i * 2 + 0] = xw[i * 2 + 0] * w + bg[0];
    g[i * 2 + 1] = xw[i * 2 + 1] * w + bg[1];
  }
}
__global__ void k_scatter(const int* __restrict__ ei, const float* __restrict__ xw,
                          const float* __restrict__ dinv, float* __restrict__ g, int E) {
  int e = blockIdx.x * blockDim.x + threadIdx.x;
  if (e < E) {
    int src = ei[e], dst = ei[E + e];
    float w = dinv[src] * dinv[dst];
    atomicAdd(&g[dst * 2 + 0], xw[src * 2 + 0] * w);
    atomicAdd(&g[dst * 2 + 1], xw[src * 2 + 1] * w);
  }
}

// LSTM elementwise: gates [64,4096] (i,f,g,o); updates c (f32), writes h as bf16
__device__ __forceinline__ float sigf(float x) { return 1.0f / (1.0f + expf(-x)); }
__global__ void k_lstm_elem(const float* __restrict__ gates, float* __restrict__ c,
                            __bf16* __restrict__ hbf) {
  int idx = blockIdx.x * blockDim.x + threadIdx.x;  // 64*1024
  if (idx < Bsz * Hd) {
    int b = idx >> 10, j = idx & 1023;
    const float* gr = gates + (size_t)b * 4 * Hd;
    float ig = sigf(gr[j]);
    float fg = sigf(gr[Hd + j]);
    float gg = tanhf(gr[2 * Hd + j]);
    float og = sigf(gr[3 * Hd + j]);
    float cn = fg * c[idx] + ig * gg;
    c[idx] = cn;
    hbf[idx] = (__bf16)(og * tanhf(cn));
  }
}

// Adjacency from prediction: denormalize, dist<=10, existence mask, self loops
__global__ void k_build_adj(const float* __restrict__ pred, const float* __restrict__ stats,
                            float* __restrict__ A) {
  __shared__ float fx[Nn], fy[Nn], ex[Nn];
  int b = blockIdx.x, t = threadIdx.x;  // 256 threads
  if (t < Nn) {
    float f0 = pred[(b * Nn + t) * 2 + 0] * stats[0] + stats[2];
    float f1 = pred[(b * Nn + t) * 2 + 1] * stats[1] + stats[3];
    fx[t] = f0; fy[t] = f1;
    ex[t] = (f0 > 0.04f && f1 > 0.04f) ? 1.0f : 0.0f;
  }
  __syncthreads();
  for (int idx = t; idx < Nn * Nn; idx += 256) {
    int i = idx >> 6, j = idx & 63;
    float dx = fx[i] - fx[j], dy = fy[i] - fy[j];
    float m = (sqrtf(dx * dx + dy * dy) <= 10.0f) ? 1.0f : 0.0f;
    float a = m * ex[i] * ex[j];
    if (i == j) a = fmaxf(a, 1.0f);
    A[((size_t)b * Nn + i) * Nn + j] = a;
  }
}
__global__ void k_dense_deg(const float* __restrict__ A, float* __restrict__ dinvd) {
  int b = blockIdx.x, i = threadIdx.x;
  const float* row = A + ((size_t)b * Nn + i) * Nn;
  float s = 0.0f;
  for (int j = 0; j < Nn; ++j) s += row[j];
  dinvd[b * Nn + i] = (s > 0.0f) ? rsqrtf(s) : 0.0f;
}
__global__ void k_dense_gcn(const float* __restrict__ A, const float* __restrict__ dinvd,
                            const float* __restrict__ xw, const float* __restrict__ bg,
                            float* __restrict__ g) {
  __shared__ float sx[Nn], sy[Nn], sd[Nn];
  int b = blockIdx.x, i = threadIdx.x;
  sx[i] = xw[(b * Nn + i) * 2 + 0];
  sy[i] = xw[(b * Nn + i) * 2 + 1];
  sd[i] = dinvd[b * Nn + i];
  __syncthreads();
  const float* row = A + ((size_t)b * Nn + i) * Nn;
  float a0 = 0.0f, a1 = 0.0f;
  for (int j = 0; j < Nn; ++j) {
    float w = row[j] * sd[j];
    a0 += w * sx[j];
    a1 += w * sy[j];
  }
  g[(b * Nn + i) * 2 + 0] = a0 * sd[i] + bg[0];
  g[(b * Nn + i) * 2 + 1] = a1 * sd[i] + bg[1];
}

// ---------------- host orchestration ----------------
extern "C" void kernel_launch(void* const* d_in, const int* in_sizes, int n_in,
                              void* d_out, int out_size, void* d_ws, size_t ws_size,
                              hipStream_t stream) {
  (void)in_sizes; (void)n_in; (void)out_size; (void)ws_size;
  const float* feat   = (const float*)d_in[0];
  const int*   ei_in  = (const int*)d_in[1];
  const float* stats  = (const float*)d_in[4];
  const float* w_gcn  = (const float*)d_in[5];
  const float* b_gcn  = (const float*)d_in[6];
  const float* w_ih   = (const float*)d_in[7];
  const float* w_hh   = (const float*)d_in[8];
  const float* b_ih   = (const float*)d_in[9];
  const float* b_hh   = (const float*)d_in[10];
  const float* wmlp[6] = {(const float*)d_in[11], (const float*)d_in[13], (const float*)d_in[15],
                          (const float*)d_in[17], (const float*)d_in[19], (const float*)d_in[21]};
  const float* bmlp[6] = {(const float*)d_in[12], (const float*)d_in[14], (const float*)d_in[16],
                          (const float*)d_in[18], (const float*)d_in[20], (const float*)d_in[22]};
  float* out = (float*)d_out;

  static const int MO[6] = {2048, 4096, 2048, 1024, 512, 128};   // layer out dims
  static const int MK[6] = {1024, 2048, 4096, 2048, 1024, 512};  // layer in dims

  // workspace carve-up
  char* base = (char*)d_ws;
  size_t off = 0;
  auto carve = [&](size_t bytes) { char* p = base + off; off = (off + bytes + 255) & ~(size_t)255; return p; };
  __bf16* wb_ih = (__bf16*)carve((size_t)4 * Hd * (Fd * Nn) * 2);
  __bf16* wb_hh = (__bf16*)carve((size_t)4 * Hd * Hd * 2);
  __bf16* wb[6];
  for (int l = 0; l < 6; ++l) wb[l] = (__bf16*)carve((size_t)MO[l] * MK[l] * 2);
  float*  b_lstm = (float*)carve((size_t)4 * Hd * 4);
  __bf16* hbf    = (__bf16*)carve((size_t)Bsz * Hd * 2);
  float*  cst    = (float*)carve((size_t)Bsz * Hd * 4);
  float*  gates  = (float*)carve((size_t)Bsz * 4 * Hd * 4);
  __bf16* bufA   = (__bf16*)carve((size_t)Bsz * 4096 * 2);
  __bf16* bufB   = (__bf16*)carve((size_t)Bsz * 4096 * 2);
  float*  gvec   = (float*)carve((size_t)NT * Fd * 4);
  __bf16* gx     = (__bf16*)carve((size_t)NT * Fd * 2);
  float*  xw     = (float*)carve((size_t)NT * Fd * 4);
  float*  deg    = (float*)carve((size_t)NT * 4);
  float*  dinv   = (float*)carve((size_t)NT * 4);
  float*  Adj    = (float*)carve((size_t)Bsz * Nn * Nn * 4);
  float*  dinvd  = (float*)carve((size_t)NT * 4);

  auto conv = [&](const float* s, __bf16* d, int n) {
    int g = (n + 255) / 256; if (g > 2048) g = 2048;
    k_f32_to_bf16<<<g, 256, 0, stream>>>(s, d, n);
  };
  auto gemm_grid = [](int N) { return (N + 255) / 256; };  // 8 waves x 32 cols

  // ---- one-time (per launch) setup: bf16 weights, fused LSTM bias, zero state ----
  conv(w_ih, wb_ih, 4 * Hd * Fd * Nn);
  conv(w_hh, wb_hh, 4 * Hd * Hd);
  for (int l = 0; l < 6; ++l) conv(wmlp[l], wb[l], MO[l] * MK[l]);
  k_addvec<<<16, 256, 0, stream>>>(b_ih, b_hh, b_lstm, 4 * Hd);
  k_fill_bf16<<<256, 256, 0, stream>>>(hbf, Bsz * Hd);
  k_fill_f32<<<256, 256, 0, stream>>>(cst, 0.0f, Bsz * Hd);
  // x_enc[0] = feature_input[0]
  k_copy_f32<<<32, 256, 0, stream>>>(feat, out, FRAME);

  auto run_edge_gcn = [&](const float* x, const int* ei) {
    k_fill_f32<<<16, 256, 0, stream>>>(deg, 1.0f, NT);
    k_deg_count<<<Ed / 256, 256, 0, stream>>>(ei, deg, Ed);
    k_dinv<<<16, 256, 0, stream>>>(deg, dinv, NT);
    k_xw<<<16, 256, 0, stream>>>(x, w_gcn, xw, NT);
    k_self_init<<<16, 256, 0, stream>>>(xw, dinv, b_gcn, gvec, NT);
    k_scatter<<<Ed / 256, 256, 0, stream>>>(ei, xw, dinv, gvec, Ed);
    k_f32_to_bf16<<<32, 256, 0, stream>>>(gvec, gx, NT * Fd);
  };

  auto run_lstm_mlp = [&](float* frame) {
    // gates = gx@w_ih^T + h@w_hh^T + (b_ih+b_hh)
    k_gemm64<<<gemm_grid(4 * Hd), 256, 0, stream>>>(gx, wb_ih, Fd * Nn, hbf, wb_hh, Hd,
                                                    b_lstm, 4 * Hd, 0, gates, nullptr);
    k_lstm_elem<<<Bsz * Hd / 256, 256, 0, stream>>>(gates, cst, hbf);
    // MLP: 5 relu layers (bf16 out) + final linear layer (f32 out -> frame)
    const __bf16* in = hbf;
    __bf16* pp[2] = {bufA, bufB};
    for (int l = 0; l < 5; ++l) {
      __bf16* o = pp[l & 1];
      k_gemm64<<<gemm_grid(MO[l]), 256, 0, stream>>>(in, wb[l], MK[l], nullptr, nullptr, 0,
                                                     bmlp[l], MO[l], 1, nullptr, o);
      in = o;
    }
    k_gemm64<<<gemm_grid(MO[5]), 256, 0, stream>>>(in, wb[5], MK[5], nullptr, nullptr, 0,
                                                   bmlp[5], MO[5], 0, frame, nullptr);
  };

  // ---- encoder: 9 steps ----
  for (int i = 0; i < T_IN - 1; ++i) {
    run_edge_gcn(feat + (size_t)i * FRAME, ei_in + (size_t)i * 2 * Ed);
    run_lstm_mlp(out + (size_t)(i + 1) * FRAME);
  }
  // ---- decoder: 6 steps ----
  for (int i = 0; i < T_OUT; ++i) {
    if (i == 0) {
      run_edge_gcn(feat + (size_t)(T_IN - 1) * FRAME, ei_in + (size_t)(T_IN - 1) * 2 * Ed);
    } else {
      const float* pred = out + (size_t)(T_IN + i - 1) * FRAME;
      k_build_adj<<<Bsz, 256, 0, stream>>>(pred, stats, Adj);
      k_dense_deg<<<Bsz, Nn, 0, stream>>>(Adj, dinvd);
      k_xw<<<16, 256, 0, stream>>>(pred, w_gcn, xw, NT);
      k_dense_gcn<<<Bsz, Nn, 0, stream>>>(Adj, dinvd, xw, b_gcn, gvec);
      k_f32_to_bf16<<<32, 256, 0, stream>>>(gvec, gx, NT * Fd);
    }
    run_lstm_mlp(out + (size_t)(T_IN + i) * FRAME);
  }
}
